// SwinTransformerBlock_37847251812981
// MI455X (gfx1250) — matrix operand order repaired and checked by reading, not compiled, not run on hardware
//
#include <hip/hip_runtime.h>
#include <hip/hip_bf16.h>

// ---------------- constants (compile-time: shapes fixed by reference) -------
#define WS7      7
#define NH       3
#define DIMC     96
#define SHIFT3   3
#define HEADD    32
#define NWIN     49            // tokens per window (7*7)
#define HH       224
#define WWID     224
#define BB       16
#define TTOK     (BB * HH * WWID)          // 802816 tokens
#define NWINH    32                        // 224/7
#define BW       (BB * NWINH * NWINH)      // 16384 windows
#define MLPH     384
#define QKVN     288
#define SCALEQ   0.17677669529663689f      // 32^-0.5

typedef __attribute__((ext_vector_type(16))) _Float16 v16h;
typedef __attribute__((ext_vector_type(8)))  _Float16 v8h;
typedef __attribute__((ext_vector_type(8)))  float    v8f;

// ---------------------------------------------------------------------------
// WMMA helper:  D = A(16x32 f16) * B(32x16 f16) + C(16x16 f32)
__device__ __forceinline__ v8f wmma16(v16h a, v16h b, v8f c) {
    return __builtin_amdgcn_wmma_f32_16x16x32_f16(
        /*neg_a=*/false, a, /*neg_b=*/false, b,
        /*c_mod=*/(short)0, c, /*reuse_a=*/false, /*reuse_b=*/false);
}

// A-fragment 16x32 f16.  element i<8 -> K = k0 + hsel*8 + i ; i>=8 -> +16.
// base points at element [row0][k0] of a row-major [.. , ld] f16 matrix.
__device__ __forceinline__ v16h load_frag_a(const _Float16* base, int ld, int lane) {
    const _Float16* p = base + (size_t)(lane & 15) * ld + ((lane >> 4) & 1) * 8;
    v8h lo = *(const v8h*)p;
    v8h hi = *(const v8h*)(p + 16);
    v16h a;
#pragma unroll
    for (int i = 0; i < 8; ++i) { a[i] = lo[i]; a[i + 8] = hi[i]; }
    return a;
}

// B-fragment 32x16 f16 taken from a TRANSPOSED source srcT[n][k] (row-major,
// leading dim ld): element B[k][n] = srcT[n][k].  Lane holds column n;
// element i<8 -> K = hsel*8 + i ; i>=8 -> +16 (same order as A).
__device__ __forceinline__ v16h load_frag_b(const _Float16* baseT, int ld, int lane) {
    const _Float16* p = baseT + (size_t)(lane & 15) * ld + ((lane >> 4) & 1) * 8;
    v8h lo = *(const v8h*)p;
    v8h hi = *(const v8h*)(p + 16);
    v16h b;
#pragma unroll
    for (int i = 0; i < 8; ++i) { b[i] = lo[i]; b[i + 8] = hi[i]; }
    return b;
}

// ---------------------------------------------------------------------------
// K0a: convert fp32 weight [K][N] into f16 transposed [N][K]
__global__ void k_wconv(const float* __restrict__ w, _Float16* __restrict__ wt,
                        int K, int N) {
    int i = blockIdx.x * 256 + threadIdx.x;
    if (i < K * N) {
        int k = i / N, n = i % N;
        wt[(size_t)n * K + k] = (_Float16)w[i];
    }
}

// ---------------------------------------------------------------------------
// K0b: LayerNorm1 — one wave (32 lanes) per token, 96 dims (3 per lane)
__global__ __launch_bounds__(256) void k_ln1(const float* __restrict__ x,
                                             const float* __restrict__ g,
                                             const float* __restrict__ b,
                                             _Float16* __restrict__ xn) {
    int tok  = (blockIdx.x * blockDim.x + threadIdx.x) >> 5;
    int lane = threadIdx.x & 31;
    const float* row = x + (size_t)tok * DIMC;
    float v0 = row[lane], v1 = row[lane + 32], v2 = row[lane + 64];
    float s = v0 + v1 + v2;
#pragma unroll
    for (int m = 1; m < 32; m <<= 1) s += __shfl_xor(s, m, 32);
    float mu = s * (1.0f / 96.0f);
    float d0 = v0 - mu, d1 = v1 - mu, d2 = v2 - mu;
    float q = d0 * d0 + d1 * d1 + d2 * d2;
#pragma unroll
    for (int m = 1; m < 32; m <<= 1) q += __shfl_xor(q, m, 32);
    float rstd = rsqrtf(q * (1.0f / 96.0f) + 1e-5f);
    _Float16* o = xn + (size_t)tok * DIMC;
    o[lane]      = (_Float16)(d0 * rstd * g[lane]      + b[lane]);
    o[lane + 32] = (_Float16)(d1 * rstd * g[lane + 32] + b[lane + 32]);
    o[lane + 64] = (_Float16)(d2 * rstd * g[lane + 64] + b[lane + 64]);
}

// ---------------------------------------------------------------------------
// K1: QKV GEMM  [T x 96] x [96 x 288] + bias -> f16 [T x 288] (token order)
__global__ __launch_bounds__(256) void k_qkv(const _Float16* __restrict__ xn,
                                             const _Float16* __restrict__ wT, // [288][96]
                                             const float* __restrict__ bias,
                                             _Float16* __restrict__ qkv) {
    __shared__ _Float16 As[64 * DIMC];          // 12 KB
    int row0 = blockIdx.x * 64;
    { // cooperative 64x96 f16 tile load (16B chunks)
        const uint4* src = (const uint4*)(xn + (size_t)row0 * DIMC);
        uint4* dst = (uint4*)As;
        for (int i = threadIdx.x; i < 64 * DIMC / 8; i += 256) dst[i] = src[i];
    }
    __syncthreads();
    int wv = threadIdx.x >> 5, lane = threadIdx.x & 31;
    int mrow  = (wv & 3) * 16;                  // 4 M-subtiles
    int nbase = (wv >> 2) * 9;                  // waves 0-3: N-tiles 0..8, 4-7: 9..17
    v16h a0 = load_frag_a(As + mrow * DIMC +  0, DIMC, lane);
    v16h a1 = load_frag_a(As + mrow * DIMC + 32, DIMC, lane);
    v16h a2 = load_frag_a(As + mrow * DIMC + 64, DIMC, lane);
    for (int nt = 0; nt < 9; ++nt) {
        int n0 = (nbase + nt) * 16;
        // prefetch next tile's weight rows into cache (global_prefetch_b8)
        if (nt < 8)
            __builtin_prefetch(wT + (size_t)(n0 + 16) * DIMC + (lane & 15) * DIMC, 0, 3);
        v16h b0 = load_frag_b(wT + (size_t)n0 * DIMC +  0, DIMC, lane);
        v16h b1 = load_frag_b(wT + (size_t)n0 * DIMC + 32, DIMC, lane);
        v16h b2 = load_frag_b(wT + (size_t)n0 * DIMC + 64, DIMC, lane);
        v8f c = {};
        c = wmma16(a0, b0, c);
        c = wmma16(a1, b1, c);
        c = wmma16(a2, b2, c);
        int col = n0 + (lane & 15);
        float bv = bias[col];
        int rbase = row0 + mrow + ((lane >> 4) ? 8 : 0);
#pragma unroll
        for (int r = 0; r < 8; ++r)
            qkv[(size_t)(rbase + r) * QKVN + col] = (_Float16)(c[r] + bv);
    }
}

// ---------------------------------------------------------------------------
// K2: windowed attention — one wave per (window, head)
__global__ __launch_bounds__(32) void k_attn(const _Float16* __restrict__ qkv,
                                             const float* __restrict__ rpb,   // [169][3]
                                             float* __restrict__ attnw,
                                             _Float16* __restrict__ aout) {   // [Bw*49][96]
    __shared__ _Float16 Qs[64 * 32];   // rows n (query), cols d
    __shared__ _Float16 Ks[64 * 32];   // rows m (key),   cols d
    __shared__ _Float16 Vt[32 * 64];   // transposed: rows d, cols n (key)
    __shared__ _Float16 Pb[64 * 64];   // softmax probs f16

    int bid  = blockIdx.x;
    int win  = bid / NH, head = bid % NH;
    int lane = threadIdx.x;
    int b  = win >> 10;
    int wi = (win >> 5) & 31;
    int wj = win & 31;

    { // zero pad rows/cols of Qs/Ks/Vt
        for (int i = lane; i < 64 * 32 / 2; i += 32) {
            ((uint*)Qs)[i] = 0u; ((uint*)Ks)[i] = 0u; ((uint*)Vt)[i] = 0u;
        }
    }
    __syncthreads();

    // gather Q,K,V rows (shift-roll is an index permutation; no pad since 224%7==0)
    for (int n = lane; n < NWIN; n += 32) {
        int rr = n / WS7, cc = n % WS7;
        int h = wi * WS7 + rr + SHIFT3; if (h >= HH)   h -= HH;
        int w = wj * WS7 + cc + SHIFT3; if (w >= WWID) w -= WWID;
        size_t t = ((size_t)(b * HH + h) * WWID + w);
        const uint4* q4 = (const uint4*)(qkv + t * QKVN + head * HEADD);
        const uint4* k4 = (const uint4*)(qkv + t * QKVN + DIMC + head * HEADD);
        uint4* qs = (uint4*)(Qs + n * 32);
        uint4* ks = (uint4*)(Ks + n * 32);
        qs[0] = q4[0]; qs[1] = q4[1]; qs[2] = q4[2]; qs[3] = q4[3];
        ks[0] = k4[0]; ks[1] = k4[1]; ks[2] = k4[2]; ks[3] = k4[3];
        // V: vector loads (4 x b128), then transposed scatter into LDS
        const v8h* vv8 = (const v8h*)(qkv + t * QKVN + 2 * DIMC + head * HEADD);
        v8h p0 = vv8[0], p1 = vv8[1], p2 = vv8[2], p3 = vv8[3];
#pragma unroll
        for (int d = 0; d < 8; ++d) {
            Vt[(d +  0) * 64 + n] = p0[d];
            Vt[(d +  8) * 64 + n] = p1[d];
            Vt[(d + 16) * 64 + n] = p2[d];
            Vt[(d + 24) * 64 + n] = p3[d];
        }
    }
    __syncthreads();

    int col_in = lane & 15;
    int rowsel = (lane >> 4) ? 8 : 0;

    // S = (Q*scale) K^T + bias ; softmax ; P -> LDS + attn_weights -> global
    for (int nt = 0; nt < 4; ++nt) {
        v16h aq = load_frag_a(Qs + nt * 16 * 32, 32, lane);
        v8f acc[4];
#pragma unroll
        for (int mt = 0; mt < 4; ++mt) {
            v16h bk = load_frag_b(Ks + mt * 16 * 32, 32, lane); // B[d][m]=K[m][d]
            v8f z = {};
            acc[mt] = wmma16(aq, bk, z);
        }
        int rowoff = nt * 16 + rowsel;
        // scale + relative-position bias + OOB mask
#pragma unroll
        for (int mt = 0; mt < 4; ++mt) {
            int m = mt * 16 + col_in;
#pragma unroll
            for (int r = 0; r < 8; ++r) {
                int n = rowoff + r;
                float s;
                if (m < NWIN && n < NWIN) {
                    int r1 = n / WS7, c1 = n % WS7, r2 = m / WS7, c2 = m % WS7;
                    int idx = (r1 - r2 + WS7 - 1) * (2 * WS7 - 1) + (c1 - c2 + WS7 - 1);
                    s = acc[mt][r] * SCALEQ + rpb[idx * NH + head];
                } else {
                    s = -1e30f;
                }
                acc[mt][r] = s;
            }
        }
        // row softmax (16-lane groups hold one full row per m-tile)
#pragma unroll
        for (int r = 0; r < 8; ++r) {
            float mx = fmaxf(fmaxf(acc[0][r], acc[1][r]), fmaxf(acc[2][r], acc[3][r]));
#pragma unroll
            for (int msk = 1; msk < 16; msk <<= 1) mx = fmaxf(mx, __shfl_xor(mx, msk, 32));
            float e0 = __expf(acc[0][r] - mx);
            float e1 = __expf(acc[1][r] - mx);
            float e2 = __expf(acc[2][r] - mx);
            float e3 = __expf(acc[3][r] - mx);
            float sm = e0 + e1 + e2 + e3;
#pragma unroll
            for (int msk = 1; msk < 16; msk <<= 1) sm += __shfl_xor(sm, msk, 32);
            float inv = 1.0f / sm;
            acc[0][r] = e0 * inv; acc[1][r] = e1 * inv;
            acc[2][r] = e2 * inv; acc[3][r] = e3 * inv;
        }
        // store attn_weights (fp32) + P (f16 to LDS)
#pragma unroll
        for (int mt = 0; mt < 4; ++mt) {
            int m = mt * 16 + col_in;
#pragma unroll
            for (int r = 0; r < 8; ++r) {
                int n = rowoff + r;
                float p = acc[mt][r];
                if (n < NWIN && m < NWIN)
                    attnw[((size_t)bid * NWIN + n) * NWIN + m] = p;
                Pb[n * 64 + m] = (_Float16)p;
            }
        }
    }
    __syncthreads();

    // O = P V   (K = 64, 2 wmma k-steps; V padding cols are zero)
    v16h vf[2][2];
#pragma unroll
    for (int kt = 0; kt < 2; ++kt)
#pragma unroll
        for (int dt = 0; dt < 2; ++dt)
            vf[kt][dt] = load_frag_b(Vt + dt * 16 * 64 + kt * 32, 64, lane); // B[n][d]=Vt[d][n]
    for (int nt = 0; nt < 4; ++nt) {
        v8f o0 = {}, o1 = {};
#pragma unroll
        for (int kt = 0; kt < 2; ++kt) {
            v16h ap = load_frag_a(Pb + nt * 16 * 64 + kt * 32, 64, lane);
            o0 = wmma16(ap, vf[kt][0], o0);
            o1 = wmma16(ap, vf[kt][1], o1);
        }
        int rowoff = nt * 16 + rowsel;
#pragma unroll
        for (int r = 0; r < 8; ++r) {
            int n = rowoff + r;
            if (n < NWIN) {
                size_t rowi = (size_t)win * NWIN + n;
                aout[rowi * DIMC + head * HEADD + col_in]      = (_Float16)o0[r];
                aout[rowi * DIMC + head * HEADD + 16 + col_in] = (_Float16)o1[r];
            }
        }
    }
}

// ---------------------------------------------------------------------------
// K3: proj GEMM + window-reverse scatter + residual  -> y1 (fp32, token order)
__global__ __launch_bounds__(256) void k_proj(const _Float16* __restrict__ aout,
                                              const _Float16* __restrict__ wT, // [96][96]
                                              const float* __restrict__ bias,
                                              const float* __restrict__ x,
                                              float* __restrict__ y1) {
    __shared__ _Float16 As[64 * DIMC];
    int row0 = blockIdx.x * 64;  // windowed row index (win*49+n)
    {
        const uint4* src = (const uint4*)(aout + (size_t)row0 * DIMC);
        uint4* dst = (uint4*)As;
        for (int i = threadIdx.x; i < 64 * DIMC / 8; i += 256) dst[i] = src[i];
    }
    __syncthreads();
    int wv = threadIdx.x >> 5, lane = threadIdx.x & 31;
    for (int t3 = 0; t3 < 3; ++t3) {
        int tile = wv * 3 + t3;                  // 24 tiles: 4 M x 6 N
        int mt = tile / 6, ntile = tile % 6;
        v16h a0 = load_frag_a(As + mt * 16 * DIMC +  0, DIMC, lane);
        v16h a1 = load_frag_a(As + mt * 16 * DIMC + 32, DIMC, lane);
        v16h a2 = load_frag_a(As + mt * 16 * DIMC + 64, DIMC, lane);
        v16h b0 = load_frag_b(wT + (size_t)ntile * 16 * DIMC +  0, DIMC, lane);
        v16h b1 = load_frag_b(wT + (size_t)ntile * 16 * DIMC + 32, DIMC, lane);
        v16h b2 = load_frag_b(wT + (size_t)ntile * 16 * DIMC + 64, DIMC, lane);
        v8f c = {};
        c = wmma16(a0, b0, c);
        c = wmma16(a1, b1, c);
        c = wmma16(a2, b2, c);
        int col = ntile * 16 + (lane & 15);
        float bv = bias[col];
        int rbase = row0 + mt * 16 + ((lane >> 4) ? 8 : 0);
#pragma unroll
        for (int r = 0; r < 8; ++r) {
            int wrow = rbase + r;
            int win = wrow / NWIN, n = wrow % NWIN;
            int b_ = win >> 10, wi = (win >> 5) & 31, wj = win & 31;
            int rr = n / WS7, cc = n % WS7;
            int h = wi * WS7 + rr + SHIFT3; if (h >= HH)   h -= HH;
            int w = wj * WS7 + cc + SHIFT3; if (w >= WWID) w -= WWID;
            size_t o = ((size_t)(b_ * HH + h) * WWID + w) * DIMC + col;
            y1[o] = c[r] + bv + x[o];
        }
    }
}

// ---------------------------------------------------------------------------
// K4: LN2 + fc1 + exact GELU + fc2 + residual  -> y (fp32 output)
__global__ __launch_bounds__(256) void k_mlp(const float* __restrict__ y1,
                                             const float* __restrict__ g,
                                             const float* __restrict__ be,
                                             const _Float16* __restrict__ w1T, // [384][96]
                                             const float* __restrict__ b1,
                                             const _Float16* __restrict__ w2T, // [96][384]
                                             const float* __restrict__ b2,
                                             float* __restrict__ y) {
    __shared__ _Float16 As[64 * DIMC];   // 12 KB  (LN2 output, f16)
    __shared__ _Float16 Hs[64 * MLPH];   // 48 KB  (gelu(fc1) activations)
    int row0 = blockIdx.x * 64;
    int wv = threadIdx.x >> 5, lane = threadIdx.x & 31;

    // LN2: 8 waves x 8 rows
    for (int i = 0; i < 8; ++i) {
        int r = wv * 8 + i;
        const float* xr = y1 + (size_t)(row0 + r) * DIMC;
        float v0 = xr[lane], v1 = xr[lane + 32], v2 = xr[lane + 64];
        float s = v0 + v1 + v2;
#pragma unroll
        for (int m = 1; m < 32; m <<= 1) s += __shfl_xor(s, m, 32);
        float mu = s * (1.0f / 96.0f);
        float d0 = v0 - mu, d1 = v1 - mu, d2 = v2 - mu;
        float q = d0 * d0 + d1 * d1 + d2 * d2;
#pragma unroll
        for (int m = 1; m < 32; m <<= 1) q += __shfl_xor(q, m, 32);
        float rstd = rsqrtf(q * (1.0f / 96.0f) + 1e-5f);
        As[r * DIMC + lane]      = (_Float16)(d0 * rstd * g[lane]      + be[lane]);
        As[r * DIMC + lane + 32] = (_Float16)(d1 * rstd * g[lane + 32] + be[lane + 32]);
        As[r * DIMC + lane + 64] = (_Float16)(d2 * rstd * g[lane + 64] + be[lane + 64]);
    }
    __syncthreads();

    // fc1 + GELU: 96 wave-tiles (4 M x 24 N), 12 per wave, K=96 (3 steps)
    for (int t12 = 0; t12 < 12; ++t12) {
        int tile = wv * 12 + t12;
        int mt = tile / 24, ntile = tile % 24;
        if (t12 < 11)
            __builtin_prefetch(w1T + (size_t)((tile + 1) % 24) * 16 * DIMC
                               + (lane & 15) * DIMC, 0, 3);
        v16h a0 = load_frag_a(As + mt * 16 * DIMC +  0, DIMC, lane);
        v16h a1 = load_frag_a(As + mt * 16 * DIMC + 32, DIMC, lane);
        v16h a2 = load_frag_a(As + mt * 16 * DIMC + 64, DIMC, lane);
        v16h b0 = load_frag_b(w1T + (size_t)ntile * 16 * DIMC +  0, DIMC, lane);
        v16h b1 = load_frag_b(w1T + (size_t)ntile * 16 * DIMC + 32, DIMC, lane);
        v16h b2 = load_frag_b(w1T + (size_t)ntile * 16 * DIMC + 64, DIMC, lane);
        v8f c = {};
        c = wmma16(a0, b0, c);
        c = wmma16(a1, b1, c);
        c = wmma16(a2, b2, c);
        int col = ntile * 16 + (lane & 15);
        float bv = b1[col];
        int rb = mt * 16 + ((lane >> 4) ? 8 : 0);
#pragma unroll
        for (int r = 0; r < 8; ++r) {
            float v = c[r] + bv;
            v = 0.5f * v * (1.0f + erff(v * 0.70710678118f));   // exact GELU
            Hs[(rb + r) * MLPH + col] = (_Float16)v;
        }
    }
    __syncthreads();

    // fc2 + residual: 24 wave-tiles (4 M x 6 N), 3 per wave, K=384 (12 steps)
    for (int t3 = 0; t3 < 3; ++t3) {
        int tile = wv * 3 + t3;
        int mt = tile / 6, ntile = tile % 6;
        v8f c = {};
        for (int k = 0; k < 12; ++k) {
            v16h a = load_frag_a(Hs + mt * 16 * MLPH + k * 32, MLPH, lane);
            v16h b = load_frag_b(w2T + (size_t)ntile * 16 * MLPH + k * 32, MLPH, lane);
            c = wmma16(a, b, c);
        }
        int col = ntile * 16 + (lane & 15);
        float bv = b2[col];
        int rbase = row0 + mt * 16 + ((lane >> 4) ? 8 : 0);
#pragma unroll
        for (int r = 0; r < 8; ++r) {
            size_t o = (size_t)(rbase + r) * DIMC + col;
            y[o] = c[r] + bv + y1[o];
        }
    }
}

// ---------------------------------------------------------------------------
extern "C" void kernel_launch(void* const* d_in, const int* in_sizes, int n_in,
                              void* d_out, int out_size, void* d_ws, size_t ws_size,
                              hipStream_t stream) {
    (void)in_sizes; (void)n_in; (void)out_size; (void)ws_size;
    const float* x      = (const float*)d_in[0];
    const float* w_qkv  = (const float*)d_in[1];
    const float* b_qkv  = (const float*)d_in[2];
    const float* w_proj = (const float*)d_in[3];
    const float* b_proj = (const float*)d_in[4];
    const float* rpb    = (const float*)d_in[5];
    const float* gamma1 = (const float*)d_in[6];
    const float* beta1  = (const float*)d_in[7];
    const float* gamma2 = (const float*)d_in[8];
    const float* beta2  = (const float*)d_in[9];
    const float* w_fc1  = (const float*)d_in[10];
    const float* b_fc1  = (const float*)d_in[11];
    const float* w_fc2  = (const float*)d_in[12];
    const float* b_fc2  = (const float*)d_in[13];

    float* y     = (float*)d_out;
    float* attnw = y + (size_t)TTOK * DIMC;          // second tuple output

    // workspace layout (regions overlaid when lifetimes are disjoint):
    //   [0, 1MB)        : f16 transposed weights
    //   [offA, +154MB)  : xn (f16, LN1 out)  -> reused as aout (f16 attn out)
    //   [offB, +462MB)  : qkv (f16)          -> reused as y1 (fp32)
    char* ws = (char*)d_ws;
    _Float16* wqkvT = (_Float16*)ws;                       // 288*96
    _Float16* wprojT = wqkvT + QKVN * DIMC;                // 96*96
    _Float16* wfc1T  = wprojT + DIMC * DIMC;               // 384*96
    _Float16* wfc2T  = wfc1T + MLPH * DIMC;                // 96*384
    size_t offA = (size_t)1 << 20;
    size_t offB = offA + (size_t)TTOK * DIMC * sizeof(_Float16);
    _Float16* xn   = (_Float16*)(ws + offA);
    _Float16* aout = (_Float16*)(ws + offA);
    _Float16* qkv  = (_Float16*)(ws + offB);
    float*    y1   = (float*)(ws + offB);

    // weight conversion / transposition (f16)
    k_wconv<<<(DIMC * QKVN + 255) / 256, 256, 0, stream>>>(w_qkv, wqkvT, DIMC, QKVN);
    k_wconv<<<(DIMC * DIMC + 255) / 256, 256, 0, stream>>>(w_proj, wprojT, DIMC, DIMC);
    k_wconv<<<(DIMC * MLPH + 255) / 256, 256, 0, stream>>>(w_fc1, wfc1T, DIMC, MLPH);
    k_wconv<<<(MLPH * DIMC + 255) / 256, 256, 0, stream>>>(w_fc2, wfc2T, MLPH, DIMC);

    k_ln1 <<<TTOK / 8, 256, 0, stream>>>(x, gamma1, beta1, xn);
    k_qkv <<<TTOK / 64, 256, 0, stream>>>(xn, wqkvT, b_qkv, qkv);
    k_attn<<<BW * NH, 32, 0, stream>>>(qkv, rpb, attnw, aout);
    k_proj<<<TTOK / 64, 256, 0, stream>>>(aout, wprojT, b_proj, x, y1);
    k_mlp <<<TTOK / 64, 256, 0, stream>>>(y1, gamma2, beta2, wfc1T, b_fc1,
                                          wfc2T, b_fc2, y);
}
